// ResNet_cifar10_69260642615464
// MI455X (gfx1250) — 16-bit and 8-bit matrix operand order repaired and checked by reading, compiled, not run on hardware
//
#include <hip/hip_runtime.h>

typedef __attribute__((ext_vector_type(8)))  int      v8i;
typedef __attribute__((ext_vector_type(8)))  float    v8f;
typedef __attribute__((ext_vector_type(16))) _Float16 v16h;

__device__ __forceinline__ int klocalB8(int v, int g, int b) {
  return ((v >> 2) << 5) + (g << 4) + ((v & 3) << 2) + b;
}
__device__ __forceinline__ int klocalA16(int v, int g, int h) {
  return ((v >> 2) << 4) + ((v & 3) << 1) + (g << 3) + h;
}
__device__ __forceinline__ int klocalB16(int v, int g, int h) {
  return (g << 4) + (v << 1) + h;
}
__device__ __forceinline__ signed char sgn8(float x) {
  return (x > 0.f) ? (signed char)1 : (x < 0.f ? (signed char)-1 : (signed char)0);
}
__device__ __forceinline__ int clampi(int x, int lo, int hi) {
  return x < lo ? lo : (x > hi ? hi : x);
}

__global__ void pack_wgt_i8(const float* __restrict__ w, signed char* __restrict__ out,
                            int N, int C, int ksz) {
  const int taps = ksz * ksz;
  const int KT = (C + 63) >> 6;
  const int NT = (N + 15) >> 4;
  const int tile = blockIdx.x;
  const int nt  = tile % NT;
  const int t2  = tile / NT;
  const int kt  = t2 % KT;
  const int tap = t2 / KT;
  if (tap >= taps) return;
  const int lane = threadIdx.x & 31;
  const int g = lane >> 4;
  const int n = nt * 16 + (lane & 15);
  signed char* dst = out + ((size_t)tile << 10) + lane * 32;
  for (int v = 0; v < 8; ++v)
    for (int b = 0; b < 4; ++b) {
      int k = kt * 64 + klocalB8(v, g, b);
      signed char s = 0;
      if (n < N && k < C)
        s = sgn8(w[((size_t)n * C + k) * taps + tap]);
      dst[v * 4 + b] = s;
    }
}

__global__ void pack_wgt_f16_conv1(const float* __restrict__ w, _Float16* __restrict__ out) {
  const int nt = blockIdx.x;
  const int lane = threadIdx.x & 31;
  const int g = lane >> 4;
  const int n = nt * 16 + (lane & 15);
  _Float16* dst = out + (size_t)nt * 512 + lane * 16;
  for (int v = 0; v < 8; ++v)
    for (int h = 0; h < 2; ++h) {
      int k = klocalA16(v, g, h);
      float s = 0.f;
      if (k < 27) s = (float)sgn8(w[n * 27 + k]);
      dst[v * 2 + h] = (_Float16)s;
    }
}

__global__ void conv1_f16_wmma(const float* __restrict__ x, const _Float16* __restrict__ wp,
                               float* __restrict__ out, float* __restrict__ stats) {
  __shared__ float ls[160];
  for (int i = threadIdx.x; i < 160; i += blockDim.x) ls[i] = 0.f;
  __syncthreads();

  const int NT = 5;
  const int wave = blockIdx.x * (blockDim.x >> 5) + (threadIdx.x >> 5);
  const int tileN = wave % NT;
  const int tM = wave / NT;
  const int lane = threadIdx.x & 31;
  const int g = lane >> 4;
  const int n = tileN * 16 + (lane & 15);
  const v16h bm = *(const v16h*)(wp + (size_t)tileN * 512 + lane * 16);

  float s1 = 0.f, s2 = 0.f;
  #pragma unroll
  for (int s = 0; s < 4; ++s) {
    const int m = tM * 64 + s * 16 + (lane & 15);
    const int b = m >> 10, yo = (m >> 5) & 31, xo = m & 31;
    v16h a;
    #pragma unroll
    for (int e = 0; e < 16; ++e) {
      const int v = e >> 1, h = e & 1;
      const int k = klocalA16(v, g, h);
      const int c = k / 9, tap = k % 9;
      const int yi = yo + tap / 3 - 1, xi = xo + (tap % 3) - 1;
      const bool valid = (k < 27) & ((unsigned)yi < 32u) & ((unsigned)xi < 32u);
      const int cc = clampi(c, 0, 2), yic = clampi(yi, 0, 31), xic = clampi(xi, 0, 31);
      const float val = x[(((size_t)b * 3 + cc) << 10) + (yic << 5) + xic];
      a[e] = (_Float16)(valid ? val : 0.f);
    }
    v8f acc = {};
    acc = __builtin_amdgcn_wmma_f32_16x16x32_f16(false, a, false, bm, (short)0, acc,
                                                 false, false);
    #pragma unroll
    for (int r = 0; r < 8; ++r) {
      const int mm = tM * 64 + s * 16 + r + (g << 3);
      const float val = acc[r];
      out[(size_t)mm * 80 + n] = val;
      s1 += val; s2 += val * val;
    }
  }
  atomicAdd(&ls[n], s1);
  atomicAdd(&ls[80 + n], s2);
  __syncthreads();
  for (int i = threadIdx.x; i < 160; i += blockDim.x) atomicAdd(&stats[i], ls[i]);
}

__global__ void bconv_i8_wmma(const signed char* __restrict__ act,
                              const signed char* __restrict__ wp,
                              const float* __restrict__ resid,
                              float* __restrict__ out, float* __restrict__ stats,
                              int Hin, int Win, int H, int W,
                              int Cpad, int Cout, int stride, int ksz, int pad) {
  __shared__ float ls[640];
  for (int i = threadIdx.x; i < 2 * Cout; i += blockDim.x) ls[i] = 0.f;
  __syncthreads();

  const int taps = ksz * ksz;
  const int KT = Cpad >> 6;
  const int NT = Cout >> 4;
  const int wave = blockIdx.x * (blockDim.x >> 5) + (threadIdx.x >> 5);
  const int tileN = wave % NT;
  const int tM = wave / NT;
  const int lane = threadIdx.x & 31;
  const int g = lane >> 4;
  const int hw = H * W;

  int bA[4], yoA[4], xoA[4];
  #pragma unroll
  for (int s = 0; s < 4; ++s) {
    const int m = tM * 64 + s * 16 + (lane & 15);
    const int bb = m / hw;
    const int rem = m - bb * hw;
    bA[s] = bb; yoA[s] = rem / W; xoA[s] = rem - (rem / W) * W;
  }

  v8i acc[4] = {};
  for (int tap = 0; tap < taps; ++tap) {
    const int dy = tap / ksz - pad, dx = tap % ksz - pad;
    const signed char* arow[4];
    int amask[4];
    #pragma unroll
    for (int s = 0; s < 4; ++s) {
      const int yi = yoA[s] * stride + dy, xi = xoA[s] * stride + dx;
      const bool inb = ((unsigned)yi < (unsigned)Hin) && ((unsigned)xi < (unsigned)Win);
      const int yic = clampi(yi, 0, Hin - 1), xic = clampi(xi, 0, Win - 1);
      arow[s] = act + (size_t)((bA[s] * Hin + yic) * Win + xic) * Cpad;
      amask[s] = inb ? ~0 : 0;
    }
    for (int kt = 0; kt < KT; ++kt) {
      const signed char* bbase =
          wp + ((size_t)((tap * KT + kt) * NT + tileN) << 10) + lane * 32;
      __builtin_prefetch(bbase + (size_t)NT * 1024, 0, 1);
      v8i bv;
      {
        const int4 t0 = *(const int4*)bbase;
        const int4 t1 = *(const int4*)(bbase + 16);
        bv[0] = t0.x; bv[1] = t0.y; bv[2] = t0.z; bv[3] = t0.w;
        bv[4] = t1.x; bv[5] = t1.y; bv[6] = t1.z; bv[7] = t1.w;
      }
      #pragma unroll
      for (int s = 0; s < 4; ++s) {
        const signed char* abase = arow[s] + (kt << 6);
        v8i av;
        #pragma unroll
        for (int p = 0; p < 4; ++p) {
          const int2 t = *(const int2*)(abase + ((p >> 1) << 5) + (g << 4) + ((p & 1) << 3));
          av[2 * p]     = t.x & amask[s];
          av[2 * p + 1] = t.y & amask[s];
        }
        acc[s] = __builtin_amdgcn_wmma_i32_16x16x64_iu8(true, av, true, bv, acc[s],
                                                        false, false);
      }
    }
  }

  const int n0 = tileN * 16 + (lane & 15);
  float s1 = 0.f, s2 = 0.f;
  #pragma unroll
  for (int s = 0; s < 4; ++s)
    #pragma unroll
    for (int r = 0; r < 8; ++r) {
      const int mm = tM * 64 + s * 16 + r + (g << 3);
      float val = (float)acc[s][r];
      if (resid) val += resid[(size_t)mm * Cout + n0];
      out[(size_t)mm * Cout + n0] = val;
      s1 += val; s2 += val * val;
    }
  atomicAdd(&ls[n0], s1);
  atomicAdd(&ls[Cout + n0], s2);
  __syncthreads();
  for (int i = threadIdx.x; i < 2 * Cout; i += blockDim.x) atomicAdd(&stats[i], ls[i]);
}

__global__ void bn_finalize(float* __restrict__ stats, const float* __restrict__ gamma,
                            const float* __restrict__ beta, int Npix, int C) {
  const int c = blockIdx.x * blockDim.x + threadIdx.x;
  if (c >= C) return;
  const float cnt = (float)Npix;
  const float mean = stats[c] / cnt;
  const float var = stats[C + c] / cnt - mean * mean;
  const float sc = gamma[c] * rsqrtf(var + 1e-5f);
  stats[2 * C + c] = sc;
  stats[3 * C + c] = beta[c] - mean * sc;
}

__global__ void bn_apply4(const float* __restrict__ in, const float* __restrict__ stats,
                          float* __restrict__ fout, signed char* __restrict__ pout,
                          int Npix, int C, int Cpad, int doRelu) {
  const size_t i4 = (size_t)blockIdx.x * blockDim.x + threadIdx.x;
  const int C4 = Cpad >> 2;
  const size_t total4 = (size_t)Npix * C4;
  if (i4 >= total4) return;
  const int c = ((int)(i4 % (size_t)C4)) << 2;
  const size_t pix = i4 / (size_t)C4;
  char4 pk;
  if (c >= C) {
    pk.x = pk.y = pk.z = pk.w = 0;
    if (pout) ((char4*)pout)[i4] = pk;
    return;
  }
  const float4 xv = *(const float4*)(in + pix * C + c);
  const float* sc = stats + 2 * C;
  const float* sh = stats + 3 * C;
  float y0 = fmaf(xv.x, sc[c + 0], sh[c + 0]);
  float y1 = fmaf(xv.y, sc[c + 1], sh[c + 1]);
  float y2 = fmaf(xv.z, sc[c + 2], sh[c + 2]);
  float y3 = fmaf(xv.w, sc[c + 3], sh[c + 3]);
  if (doRelu) {
    y0 = fmaxf(y0, 0.f); y1 = fmaxf(y1, 0.f);
    y2 = fmaxf(y2, 0.f); y3 = fmaxf(y3, 0.f);
  }
  if (fout) *(float4*)(fout + pix * C + c) = make_float4(y0, y1, y2, y3);
  if (pout) {
    pk.x = sgn8(y0); pk.y = sgn8(y1); pk.z = sgn8(y2); pk.w = sgn8(y3);
    ((char4*)pout)[i4] = pk;
  }
}

__global__ void avgpool_fc(const float* __restrict__ h, const float* __restrict__ fcw,
                           const float* __restrict__ fcb, float* __restrict__ out) {
  __shared__ float pooled[320];
  const int b = blockIdx.x;
  for (int c = threadIdx.x; c < 320; c += blockDim.x) {
    float s = 0.f;
    for (int p = 0; p < 64; ++p)
      s += h[((size_t)b * 64 + p) * 320 + c];
    s *= (1.f / 64.f);
    pooled[c] = (float)sgn8(s);
  }
  __syncthreads();
  if (threadIdx.x < 10) {
    const int j = threadIdx.x;
    float s = fcb[j];
    for (int c = 0; c < 320; ++c)
      s += pooled[c] * (float)sgn8(fcw[j * 320 + c]);
    out[b * 10 + j] = s;
  }
}

static inline size_t pad64(size_t c) { return (c + 63) & ~(size_t)63; }

extern "C" void kernel_launch(void* const* d_in, const int* in_sizes, int n_in,
                              void* d_out, int out_size, void* d_ws, size_t ws_size,
                              hipStream_t stream) {
  (void)in_sizes; (void)n_in; (void)out_size; (void)ws_size;

  static const int DS[6]  = {0, 0, 1, 0, 1, 0};
  static const int INP[6] = {80, 80, 80, 160, 160, 320};
  static const int PL[6]  = {80, 80, 160, 160, 320, 320};
  static const int STR[6] = {1, 1, 2, 1, 2, 1};

  int idx = 0;
  const float* X   = (const float*)d_in[idx++];
  const float* CW1 = (const float*)d_in[idx++];
  const float* G0  = (const float*)d_in[idx++];
  const float* B0  = (const float*)d_in[idx++];
  const float *w1[6], *g1[6], *b1[6], *w2[6], *g2[6], *b2[6], *dw[6], *dg[6], *db[6];
  for (int i = 0; i < 6; ++i) {
    w1[i] = (const float*)d_in[idx++]; g1[i] = (const float*)d_in[idx++];
    b1[i] = (const float*)d_in[idx++]; w2[i] = (const float*)d_in[idx++];
    g2[i] = (const float*)d_in[idx++]; b2[i] = (const float*)d_in[idx++];
    if (DS[i]) {
      dw[i] = (const float*)d_in[idx++]; dg[i] = (const float*)d_in[idx++];
      db[i] = (const float*)d_in[idx++];
    } else { dw[i] = dg[i] = db[i] = nullptr; }
  }
  const float* FCW = (const float*)d_in[idx++];
  const float* FCB = (const float*)d_in[idx++];

  char* ws = (char*)d_ws;
  size_t off = 0;
  auto walloc = [&](size_t bytes) -> void* {
    void* p = ws + off; off += (bytes + 1023) & ~(size_t)1023; return p;
  };
  float* F0 = (float*)walloc((size_t)256 * 32 * 32 * 80 * 4);
  float* F1 = (float*)walloc((size_t)256 * 32 * 32 * 80 * 4);
  float* F2 = (float*)walloc((size_t)256 * 16 * 16 * 160 * 4);
  signed char* P0 = (signed char*)walloc((size_t)256 * 32 * 32 * 128);
  signed char* P1 = (signed char*)walloc((size_t)256 * 32 * 32 * 128);
  float* STATS = (float*)walloc(8192);
  _Float16* WPC1 = (_Float16*)walloc(5 * 1024);
  signed char* WP = (signed char*)walloc(8u << 20);

  auto tilesOf = [](int N, int C, int k) -> size_t {
    return (size_t)k * k * ((C + 63) / 64) * ((N + 15) / 16);
  };
  hipLaunchKernelGGL(pack_wgt_f16_conv1, dim3(5), dim3(32), 0, stream, CW1, WPC1);

  signed char *wp1[6], *wp2[6], *wpd[6];
  size_t woff = 0;
  for (int i = 0; i < 6; ++i) {
    size_t t;
    t = tilesOf(PL[i], INP[i], 3);
    wp1[i] = WP + woff; woff += t << 10;
    hipLaunchKernelGGL(pack_wgt_i8, dim3((unsigned)t), dim3(32), 0, stream,
                       w1[i], wp1[i], PL[i], INP[i], 3);
    t = tilesOf(PL[i], PL[i], 3);
    wp2[i] = WP + woff; woff += t << 10;
    hipLaunchKernelGGL(pack_wgt_i8, dim3((unsigned)t), dim3(32), 0, stream,
                       w2[i], wp2[i], PL[i], PL[i], 3);
    if (DS[i]) {
      t = tilesOf(PL[i], INP[i], 1);
      wpd[i] = WP + woff; woff += t << 10;
      hipLaunchKernelGGL(pack_wgt_i8, dim3((unsigned)t), dim3(32), 0, stream,
                         dw[i], wpd[i], PL[i], INP[i], 1);
    } else wpd[i] = nullptr;
  }

  auto run_bconv = [&](const signed char* act, const signed char* wpk,
                       const float* resid, float* out,
                       int Hin, int Win, int H, int W, int Cpad, int Cout,
                       int stride, int ksz, int pad) {
    hipMemsetAsync(STATS, 0, 2 * Cout * sizeof(float), stream);
    const int waves = ((256 * H * W) >> 6) * (Cout >> 4);
    hipLaunchKernelGGL(bconv_i8_wmma, dim3(waves / 8), dim3(256), 0, stream,
                       act, wpk, resid, out, STATS,
                       Hin, Win, H, W, Cpad, Cout, stride, ksz, pad);
  };
  auto run_apply = [&](const float* in, const float* g, const float* b,
                       float* fout, signed char* pout, int Npix, int C, int Cpad, int relu) {
    hipLaunchKernelGGL(bn_finalize, dim3((C + 255) / 256), dim3(256), 0, stream,
                       STATS, g, b, Npix, C);
    const size_t total4 = (size_t)Npix * (Cpad >> 2);
    hipLaunchKernelGGL(bn_apply4, dim3((unsigned)((total4 + 255) / 256)), dim3(256), 0,
                       stream, in, STATS, fout, pout, Npix, C, Cpad, relu);
  };

  {
    hipMemsetAsync(STATS, 0, 2 * 80 * sizeof(float), stream);
    const int waves = (262144 >> 6) * 5;
    hipLaunchKernelGGL(conv1_f16_wmma, dim3(waves / 8), dim3(256), 0, stream,
                       X, WPC1, F1, STATS);
    run_apply(F1, G0, B0, F0, P0, 256 * 32 * 32, 80, 128, 1);
  }

  int Hc = 32, Wc = 32, Cpadc = 128;
  for (int i = 0; i < 6; ++i) {
    const int pl = PL[i], stride = STR[i];
    const int Ho = Hc / stride, Wo = Wc / stride;
    const int Npo = 256 * Ho * Wo;
    const int CpadO = (int)pad64(pl);

    const float* resid;
    if (DS[i]) {
      run_bconv(P0, wpd[i], nullptr, F2, Hc, Wc, Ho, Wo, Cpadc, pl, stride, 1, 0);
      run_apply(F2, dg[i], db[i], F2, nullptr, Npo, pl, pl, 0);
      resid = F2;
    } else resid = F0;

    run_bconv(P0, wp1[i], nullptr, F1, Hc, Wc, Ho, Wo, Cpadc, pl, stride, 3, 1);
    run_apply(F1, g1[i], b1[i], nullptr, P1, Npo, pl, CpadO, 1);

    run_bconv(P1, wp2[i], resid, F1, Ho, Wo, Ho, Wo, CpadO, pl, 1, 3, 1);
    run_apply(F1, g2[i], b2[i], F0, P0, Npo, pl, CpadO, 1);

    Hc = Ho; Wc = Wo; Cpadc = CpadO;
  }

  hipLaunchKernelGGL(avgpool_fc, dim3(256), dim3(256), 0, stream,
                     F0, FCW, FCB, (float*)d_out);
}
